// OnetoManyGRU_1219770712430
// MI455X (gfx1250) — matrix operand order repaired and checked by reading, not compiled, not run on hardware
//
#include <hip/hip_runtime.h>
#include <hip/hip_bf16.h>

// ---------------------------------------------------------------------------
// OnetoManyGRU for MI455X (gfx1250, wave32, WMMA)
// B=256, E=1024, D=256, T=256, K=128
// ---------------------------------------------------------------------------

typedef __attribute__((ext_vector_type(16))) __bf16 v16bf;
typedef __attribute__((ext_vector_type(8)))  float  v8f;
typedef int v4i __attribute__((vector_size(16)));   // matches builtin param type

union Frag {
  uint4 u4[2];
  v16bf v;
};

__device__ __forceinline__ unsigned short f2bf(float f) {
  union { float f; unsigned u; } v; v.f = f;
  unsigned r = v.u + 0x7FFFu + ((v.u >> 16) & 1u);   // round-to-nearest-even
  return (unsigned short)(r >> 16);
}
__device__ __forceinline__ float bf2f(unsigned short h) {
  union { unsigned u; float f; } v; v.u = ((unsigned)h) << 16;
  return v.f;
}
__device__ __forceinline__ float sigmoidf(float x) {
  return 1.0f / (1.0f + __expf(-x));
}

// ---------------------------------------------------------------------------
// Async global->LDS copy path (CDNA5 GLOBAL_LOAD_ASYNC_TO_LDS_B128, ASYNCcnt)
// with synchronous fallback if the builtin is not available.
// ---------------------------------------------------------------------------
#if defined(__has_builtin)
#if __has_builtin(__builtin_amdgcn_global_load_async_to_lds_b128)
#define HAS_ASYNC_LDS 1
#endif
#endif
#ifndef HAS_ASYNC_LDS
#define HAS_ASYNC_LDS 0
#endif

__device__ __forceinline__ void cp16(const unsigned short* g, unsigned short* l) {
#if HAS_ASYNC_LDS
  __builtin_amdgcn_global_load_async_to_lds_b128(
      (__attribute__((address_space(1))) v4i*)g,
      (__attribute__((address_space(3))) v4i*)l, 0, 0);
#else
  *(uint4*)l = *(const uint4*)g;   // global_load_b128 + ds_store_b128
#endif
}

__device__ __forceinline__ void async_join() {
#if HAS_ASYNC_LDS
#if __has_builtin(__builtin_amdgcn_s_wait_asynccnt)
  __builtin_amdgcn_s_wait_asynccnt(0);
#else
  asm volatile("s_wait_asynccnt 0x0" ::: "memory");
#endif
#endif
}

// ---------------------------------------------------------------------------
// Device-scope sense-reversal grid barrier (for the persistent GRU kernel).
// bar[0] = arrival count, bar[1] = generation. Re-zeroed every launch.
// ---------------------------------------------------------------------------
__device__ __forceinline__ void grid_barrier(unsigned* bar, unsigned nblk) {
  __syncthreads();
  if (threadIdx.x == 0) {
    __threadfence();
    const unsigned gen =
        __hip_atomic_load(bar + 1, __ATOMIC_RELAXED, __HIP_MEMORY_SCOPE_AGENT);
    if (__hip_atomic_fetch_add(bar, 1u, __ATOMIC_ACQ_REL,
                               __HIP_MEMORY_SCOPE_AGENT) == nblk - 1) {
      __hip_atomic_store(bar, 0u, __ATOMIC_RELAXED, __HIP_MEMORY_SCOPE_AGENT);
      __hip_atomic_fetch_add(bar + 1, 1u, __ATOMIC_RELEASE,
                             __HIP_MEMORY_SCOPE_AGENT);
    } else {
      while (__hip_atomic_load(bar + 1, __ATOMIC_ACQUIRE,
                               __HIP_MEMORY_SCOPE_AGENT) == gen)
        __builtin_amdgcn_s_sleep(1);
    }
    __threadfence();
  }
  __syncthreads();
}

// ---------------------------------------------------------------------------
// Generic bf16 GEMM:  C[M][N] = A[M][Kd] * W[N][Kd]^T  (+bias, epilogue EPI)
// 128 threads = 4 waves; 64x64 tile; BLOCK_K = 32; double-buffered LDS.
// EPI 0: tok  -> shifted teacher-forced inputs (bf16), row stride 1024
// EPI 1: gi   -> bf16 [32768 x 3072]
// EPI 2: untok-> f32 d_out transposed [B, D, K]
// ---------------------------------------------------------------------------
template <int EPI>
__global__ __launch_bounds__(128)
void gemm_bf16_kernel(const unsigned short* __restrict__ A,
                      const unsigned short* __restrict__ W,
                      const float* __restrict__ bias,
                      void* __restrict__ out,
                      int Kd) {
  const int t    = threadIdx.x;
  const int lane = t & 31;
  const int wave = t >> 5;
  const int mBase = blockIdx.y * 64;
  const int nBase = blockIdx.x * 64;

  __shared__ __align__(16) unsigned char smem[16384];  // sA[2]:0/4096 sB[2]:8192/12288

  v8f acc[2][2] = {};

  const int row  = t >> 1;
  const int half = t & 1;
  const unsigned short* gA = A + (size_t)(mBase + row) * Kd + half * 16;
  const unsigned short* gW = W + (size_t)(nBase + row) * Kd + half * 16;
  const int lOff = row * 32 + half * 16;

  const int mOff = (wave >> 1) * 32;
  const int nOff = (wave & 1) * 32;
  const int ml   = lane & 15;
  const int kh   = (lane >> 4) * 8;
  const int kbb  = (lane >> 4) * 16;

  auto issue = [&](int kb, int buf) {
    unsigned short* lA = (unsigned short*)(smem + buf * 4096) + lOff;
    unsigned short* lB = (unsigned short*)(smem + 8192 + buf * 4096) + lOff;
    cp16(gA + kb, lA);
    cp16(gA + kb + 8, lA + 8);
    cp16(gW + kb, lB);
    cp16(gW + kb + 8, lB + 8);
  };

  issue(0, 0);
  const int niter = Kd >> 5;
  for (int i = 0; i < niter; ++i) {
    async_join();
    __syncthreads();                       // tile i ready for all waves
    if (i + 1 < niter) issue((i + 1) * 32, (i + 1) & 1);  // prefetch next tile
    const unsigned short* sA = (const unsigned short*)(smem + (i & 1) * 4096);
    const unsigned short* sB = (const unsigned short*)(smem + 8192 + (i & 1) * 4096);

    Frag a[2], b[2];
#pragma unroll
    for (int im = 0; im < 2; ++im) {
      const int m = mOff + im * 16 + ml;
      a[im].u4[0] = *(const uint4*)&sA[m * 32 + kh];
      a[im].u4[1] = *(const uint4*)&sA[m * 32 + 16 + kh];
    }
#pragma unroll
    for (int in_ = 0; in_ < 2; ++in_) {
      const int n = nOff + in_ * 16 + ml;
      b[in_].u4[0] = *(const uint4*)&sB[n * 32 + kh];
      b[in_].u4[1] = *(const uint4*)&sB[n * 32 + 16 + kh];
    }
#pragma unroll
    for (int im = 0; im < 2; ++im)
#pragma unroll
      for (int in_ = 0; in_ < 2; ++in_)
        acc[im][in_] = __builtin_amdgcn_wmma_f32_16x16x32_bf16(
            false, a[im].v, false, b[in_].v, (short)0, acc[im][in_], false, false);
    __syncthreads();                       // all reads of tile i done
  }

  // Epilogue. C layout: lane l, reg r -> M = (l>>4)*8 + r, N = l&15.
  const int mh = (lane >> 4) * 8;
#pragma unroll
  for (int im = 0; im < 2; ++im)
#pragma unroll
    for (int in_ = 0; in_ < 2; ++in_)
#pragma unroll
      for (int r = 0; r < 8; ++r) {
        const int gm = mBase + mOff + im * 16 + mh + r;
        const int gn = nBase + nOff + in_ * 16 + ml;
        const float v = acc[im][in_][r] + bias[gn];
        if (EPI == 0) {
          // row m = b*128 + j ; shifted input row = m+1 when j < 127
          if ((gm & 127) < 127)
            ((unsigned short*)out)[((size_t)gm + 1) * 1024 + gn] = f2bf(v);
        } else if (EPI == 1) {
          ((unsigned short*)out)[(size_t)gm * 3072 + gn] = f2bf(v);
        } else {
          const int b = gm >> 7;
          const int j = gm & 127;
          ((float*)out)[((size_t)b * 256 + gn) * 128 + j] = v;
        }
      }
}

// ---------------------------------------------------------------------------
// Persistent fused GRU recurrence: all 128 steps in ONE launch.
// Grid (16,4) = 64 blocks, 384 threads = 12 waves each.
// Wave w owns (gate = w>>2, 32x32 quadrant = w&3): only 4 WMMA frags
// (32 acc VGPRs) per wave -> no VGPR-MSB pressure, good occupancy.
// Per step: gh = h @ W_hh^T (3 gates), gate math + h update fused, then
// grid barrier. W_hh (6 MB bf16) stays L2-resident for all 128 steps.
// ---------------------------------------------------------------------------
__global__ __launch_bounds__(384)
void gru_persistent_kernel(unsigned short* __restrict__ hb0,   // [B,E] bf16 ping
                           unsigned short* __restrict__ hb1,   // [B,E] bf16 pong
                           float* __restrict__ h32,            // [B,E] f32 master
                           const unsigned short* __restrict__ Whh,  // [3E,E] bf16
                           const float* __restrict__ bhh,           // [3E]
                           const unsigned short* __restrict__ gi,   // [B*K,3E] bf16
                           unsigned short* __restrict__ ys,         // [B*K,E] bf16
                           unsigned* __restrict__ bar) {
  const int t    = threadIdx.x;
  const int lane = t & 31;
  const int wave = t >> 5;        // 0..11
  const int g    = wave >> 2;     // gate 0..2
  const int quad = wave & 3;      // quadrant within 64x64 tile
  const int mOff = (quad >> 1) * 32;
  const int eOff = (quad & 1) * 32;
  const int mBase = blockIdx.y * 64;   // over B
  const int eBase = blockIdx.x * 64;   // over E

  // sA[2]: 0 / 4096 (4 KB each); sB[2]: 8192 / 20480 (12 KB each);
  // gh (3x64x64 f32 = 48 KB) reuses the whole region after the K-loop.
  __shared__ __align__(16) unsigned char smem[49152];
  float* gh = (float*)smem;

  const int ml  = lane & 15;
  const int kh  = (lane >> 4) * 8;
  const int kbb = (lane >> 4) * 16;
  const int mh  = (lane >> 4) * 8;

  // Tile-load task decomposition:
  //   A tile 64x32 bf16 = 256 x 16B  -> threads 0..255, one cp16 each
  //   W tiles 3 x 64x32 = 768 x 16B  -> all 384 threads, two cp16 each
  const int arow = t >> 2;
  const int acol = (t & 3) * 8;

  for (int k = 0; k < 128; ++k) {
    const unsigned short* hcur = (k & 1) ? hb1 : hb0;
    unsigned short*       hnxt = (k & 1) ? hb0 : hb1;

    v8f acc[2][2] = {};

    auto issue = [&](int kb, int buf) {
      if (t < 256)
        cp16(hcur + (size_t)(mBase + arow) * 1024 + kb + acol,
             (unsigned short*)(smem + buf * 4096) + arow * 32 + acol);
#pragma unroll
      for (int u = 0; u < 2; ++u) {
        const int task = t + u * 384;        // 0..767
        const int bg   = task >> 8;          // gate 0..2
        const int br   = (task & 255) >> 2;  // row 0..63
        const int bq   = (task & 3) * 8;     // col offset
        cp16(Whh + (size_t)(bg * 1024 + eBase + br) * 1024 + kb + bq,
             (unsigned short*)(smem + 8192 + buf * 12288) + bg * 2048 + br * 32 + bq);
      }
    };

    issue(0, 0);
    for (int i = 0; i < 32; ++i) {
      async_join();
      __syncthreads();                                   // tile i ready
      if (i + 1 < 32) issue((i + 1) * 32, (i + 1) & 1);  // prefetch next tile
      const unsigned short* sA = (const unsigned short*)(smem + (i & 1) * 4096);
      const unsigned short* sB =
          (const unsigned short*)(smem + 8192 + (i & 1) * 12288) + g * 2048;

      Frag a[2], b[2];
#pragma unroll
      for (int im = 0; im < 2; ++im) {
        const int m = mOff + im * 16 + ml;
        a[im].u4[0] = *(const uint4*)&sA[m * 32 + kh];
        a[im].u4[1] = *(const uint4*)&sA[m * 32 + 16 + kh];
      }
#pragma unroll
      for (int in_ = 0; in_ < 2; ++in_) {
        const int n = eOff + in_ * 16 + ml;
        b[in_].u4[0] = *(const uint4*)&sB[n * 32 + kh];
        b[in_].u4[1] = *(const uint4*)&sB[n * 32 + 16 + kh];
      }
#pragma unroll
      for (int im = 0; im < 2; ++im)
#pragma unroll
        for (int in_ = 0; in_ < 2; ++in_)
          acc[im][in_] = __builtin_amdgcn_wmma_f32_16x16x32_bf16(
              false, a[im].v, false, b[in_].v, (short)0, acc[im][in_],
              false, false);
      __syncthreads();                                   // reads of tile i done
    }

    // Exchange gate accumulators through LDS so r/z/n for one (b,e) meet.
#pragma unroll
    for (int im = 0; im < 2; ++im)
#pragma unroll
      for (int in_ = 0; in_ < 2; ++in_)
#pragma unroll
        for (int r = 0; r < 8; ++r) {
          const int lm = mOff + im * 16 + mh + r;
          const int le = eOff + in_ * 16 + ml;
          gh[g * 4096 + lm * 64 + le] = acc[im][in_][r];
        }
    __syncthreads();

    for (int idx = t; idx < 4096; idx += 384) {
      const int lm = idx >> 6, le = idx & 63;
      const int b = mBase + lm, e = eBase + le;
      const float ghr = gh[lm * 64 + le] + bhh[e];
      const float ghz = gh[4096 + lm * 64 + le] + bhh[1024 + e];
      const float ghn = gh[8192 + lm * 64 + le] + bhh[2048 + e];
      const size_t gbase = ((size_t)(b * 128 + k)) * 3072;
      const float r = sigmoidf(bf2f(gi[gbase + e]) + ghr);
      const float z = sigmoidf(bf2f(gi[gbase + 1024 + e]) + ghz);
      const float n = tanhf(bf2f(gi[gbase + 2048 + e]) + r * ghn);
      const float ho = h32[b * 1024 + e];
      const float hn = (1.0f - z) * n + z * ho;
      h32[b * 1024 + e] = hn;
      const unsigned short hb = f2bf(hn);
      hnxt[b * 1024 + e] = hb;
      ys[((size_t)(b * 128 + k)) * 1024 + e] = hb;
    }

    grid_barrier(bar, 64);   // all blocks finish step k before step k+1
  }
}

// ---------------------------------------------------------------------------
// Small helper kernels
// ---------------------------------------------------------------------------
__global__ void f32_to_bf16_kernel(const float* __restrict__ src,
                                   unsigned short* __restrict__ dst, int n) {
  const int i = blockIdx.x * blockDim.x + threadIdx.x;
  if (i < n) dst[i] = f2bf(src[i]);
}

// dst[(b*128 + j)*256 + d] = x[(b*256 + d)*256 + (128 + j)]
__global__ void build_xtf_kernel(const float* __restrict__ x,
                                 unsigned short* __restrict__ dst) {
  const int i = blockIdx.x * blockDim.x + threadIdx.x;  // B*K*D = 8388608
  if (i >= 256 * 128 * 256) return;
  const int d = i & 255;
  const int j = (i >> 8) & 127;
  const int b = i >> 15;
  dst[i] = f2bf(x[((size_t)(b * 256 + d)) * 256 + (128 + j)]);
}

// zero teacher-forced input row for step 0 of every batch
__global__ void zero_step0_kernel(unsigned short* __restrict__ inputs_bf) {
  const int i = blockIdx.x * blockDim.x + threadIdx.x;  // B*E
  if (i >= 256 * 1024) return;
  const int e = i & 1023;
  const int b = i >> 10;
  inputs_bf[((size_t)b * 128) * 1024 + e] = 0;
}

__global__ void init_h_kernel(const float* __restrict__ c,
                              float* __restrict__ h32,
                              unsigned short* __restrict__ hbf0,
                              unsigned* __restrict__ bar) {
  const int i = blockIdx.x * blockDim.x + threadIdx.x;  // B*E
  if (i < 2) bar[i] = 0;                                // reset grid barrier
  if (i >= 256 * 1024) return;
  const float v = c[i];
  h32[i] = v;
  hbf0[i] = f2bf(v);
}

// ---------------------------------------------------------------------------
extern "C" void kernel_launch(void* const* d_in, const int* in_sizes, int n_in,
                              void* d_out, int out_size, void* d_ws, size_t ws_size,
                              hipStream_t stream) {
  const float* c       = (const float*)d_in[0];
  /* d_in[1] = K (int scalar) == 128, hardcoded */
  const float* x       = (const float*)d_in[2];
  const float* W_ih    = (const float*)d_in[3];
  const float* W_hh    = (const float*)d_in[4];
  const float* b_ih    = (const float*)d_in[5];
  const float* b_hh    = (const float*)d_in[6];
  const float* W_tok   = (const float*)d_in[7];
  const float* b_tok   = (const float*)d_in[8];
  const float* W_untok = (const float*)d_in[9];
  const float* b_untok = (const float*)d_in[10];
  float* out = (float*)d_out;

  uintptr_t p = (uintptr_t)d_ws;
  auto take = [&](size_t bytes) {
    uintptr_t q = p;
    p += (bytes + 255) & ~(size_t)255;
    return q;
  };
  unsigned short* Wtok_bf   = (unsigned short*)take((size_t)1024 * 256 * 2);
  unsigned short* Wih_bf    = (unsigned short*)take((size_t)3072 * 1024 * 2);
  unsigned short* Whh_bf    = (unsigned short*)take((size_t)3072 * 1024 * 2);
  unsigned short* Wuntok_bf = (unsigned short*)take((size_t)256 * 1024 * 2);
  unsigned short* xtf_bf    = (unsigned short*)take((size_t)32768 * 256 * 2);
  unsigned short* in_bf     = (unsigned short*)take((size_t)32768 * 1024 * 2);
  unsigned short* gi_bf     = (unsigned short*)take((size_t)32768 * 3072 * 2);
  unsigned short* ys_bf     = (unsigned short*)take((size_t)32768 * 1024 * 2);
  float*          h32       = (float*)take((size_t)262144 * 4);
  unsigned short* hb0       = (unsigned short*)take((size_t)262144 * 2);
  unsigned short* hb1       = (unsigned short*)take((size_t)262144 * 2);
  unsigned*       bar       = (unsigned*)take(256);

  auto cvt = [&](const float* s, unsigned short* d, int n) {
    f32_to_bf16_kernel<<<(n + 255) / 256, 256, 0, stream>>>(s, d, n);
  };
  cvt(W_tok, Wtok_bf, 1024 * 256);
  cvt(W_ih, Wih_bf, 3072 * 1024);
  cvt(W_hh, Whh_bf, 3072 * 1024);
  cvt(W_untok, Wuntok_bf, 256 * 1024);
  build_xtf_kernel<<<(256 * 128 * 256 + 255) / 256, 256, 0, stream>>>(x, xtf_bf);
  zero_step0_kernel<<<(262144 + 255) / 256, 256, 0, stream>>>(in_bf);
  init_h_kernel<<<(262144 + 255) / 256, 256, 0, stream>>>(c, h32, hb0, bar);

  // tok: [32768x256] * W_tok[1024x256]^T -> shifted inputs (bf16)
  gemm_bf16_kernel<0><<<dim3(1024 / 64, 32768 / 64), 128, 0, stream>>>(
      xtf_bf, Wtok_bf, b_tok, (void*)in_bf, 256);
  // gi: [32768x1024] * W_ih[3072x1024]^T -> gi (bf16)
  gemm_bf16_kernel<1><<<dim3(3072 / 64, 32768 / 64), 128, 0, stream>>>(
      in_bf, Wih_bf, b_ih, (void*)gi_bf, 1024);

  // Entire 128-step recurrence in one persistent launch (12 waves/block).
  gru_persistent_kernel<<<dim3(16, 4), 384, 0, stream>>>(
      hb0, hb1, h32, Whh_bf, b_hh, gi_bf, ys_bf, bar);

  // untok: [32768x1024] * W_untok[256x1024]^T -> d_out [B, D, K] (f32)
  gemm_bf16_kernel<2><<<dim3(256 / 64, 32768 / 64), 128, 0, stream>>>(
      ys_bf, Wuntok_bf, b_untok, (void*)out, 1024);
}
